// QwenMLP_57269093925327
// MI455X (gfx1250) — compile-verified, matrix operand order checked by reading, not run on hardware
//
#include <hip/hip_runtime.h>
#include <hip/hip_bf16.h>

typedef __attribute__((ext_vector_type(16))) __bf16 v16bf;
typedef __attribute__((ext_vector_type(8)))  float  v8f;
typedef unsigned short u16;
typedef unsigned int   u32;

#define H_DIM 4096
#define I_DIM 14336
#define T_DIM 1024   // B*S = 4*256
#define RANK  4

#define BK 32        // K chunk (elements)
#define SK 40        // padded LDS row stride (halves); 80B rows -> conflict-free b128
#define BM 64        // tokens per block (fallback kernels)
#define BN 128       // output channels per block
#define NTHREADS 256 // fallback: 8 waves, 2(M) x 4(N)

#define FBM 128      // fast path: tokens per block
#define FTHREADS 512 // fast path: 16 waves, 4(M) x 4(N)

union FragBF { v16bf v; uint4 q[2]; };

__device__ __forceinline__ u16 f2bf(float f) {
    u32 u = __float_as_uint(f);
    u32 r = (u + 0x7FFFu + ((u >> 16) & 1u)) >> 16;   // round-to-nearest-even
    return (u16)r;
}

__device__ __forceinline__ uint2 pack4bf(float a, float b, float c, float d) {
    u32 lo = (u32)f2bf(a) | ((u32)f2bf(b) << 16);
    u32 hi = (u32)f2bf(c) | ((u32)f2bf(d) << 16);
    return make_uint2(lo, hi);
}

// ---- CDNA5 async copy: global -> LDS, 16B per lane, tracked by ASYNCcnt ----
__device__ __forceinline__ void async_b128(u32 lds_off, const void* sbase, u32 voff) {
    asm volatile("global_load_async_to_lds_b128 %0, %1, %2"
                 :: "v"(lds_off), "v"(voff), "s"(sbase) : "memory");
}
template <int N>
__device__ __forceinline__ void wait_async_le() {
    asm volatile("s_wait_asynccnt %0" :: "i"(N) : "memory");
}

// Load one 16x32 bf16 WMMA operand (A or B) from an LDS tile with row stride SK.
__device__ __forceinline__ v16bf load_frag(const u16* s, int row0, int lane) {
    int r  = lane & 15;
    int hi = (lane >> 4) & 1;
    const u16* p = s + (row0 + r) * SK + hi * 8;
    FragBF f;
    f.q[0] = *reinterpret_cast<const uint4*>(p);        // K 0..7  / 8..15
    f.q[1] = *reinterpret_cast<const uint4*>(p + 16);   // K 16..23/24..31
    return f.v;
}

// ============================================================================
//                    PRECOMPUTE KERNELS (fast path)
// ============================================================================

__global__ __launch_bounds__(256)
void cvt_x_kernel(const float* __restrict__ x, u16* __restrict__ xbf) {
    size_t i = ((size_t)blockIdx.x * 256 + threadIdx.x) * 4;
    float4 v = *reinterpret_cast<const float4*>(x + i);
    *reinterpret_cast<uint2*>(xbf + i) = pack4bf(v.x, v.y, v.z, v.w);
}

__global__ __launch_bounds__(256)
void scale_cvt_kernel(const float* __restrict__ w,   // [O, IN]
                      const float* __restrict__ su,  // [O, R]
                      const float* __restrict__ sv,  // [R, IN]
                      u16* __restrict__ out,         // [O, IN] bf16
                      int in_dim) {
    const int o = blockIdx.y;
    const int h = blockIdx.x * 1024 + threadIdx.x * 4;
    const float4 suv = *reinterpret_cast<const float4*>(su + (size_t)o * RANK);
    const size_t off = (size_t)o * in_dim + h;
    float4 wv = *reinterpret_cast<const float4*>(w + off);
    float4 s0 = *reinterpret_cast<const float4*>(sv + h);
    float4 s1 = *reinterpret_cast<const float4*>(sv + in_dim + h);
    float4 s2 = *reinterpret_cast<const float4*>(sv + 2 * (size_t)in_dim + h);
    float4 s3 = *reinterpret_cast<const float4*>(sv + 3 * (size_t)in_dim + h);
    float ex = wv.x * (suv.x * s0.x + suv.y * s1.x + suv.z * s2.x + suv.w * s3.x);
    float ey = wv.y * (suv.x * s0.y + suv.y * s1.y + suv.z * s2.y + suv.w * s3.y);
    float ez = wv.z * (suv.x * s0.z + suv.y * s1.z + suv.z * s2.z + suv.w * s3.z);
    float ew = wv.w * (suv.x * s0.w + suv.y * s1.w + suv.z * s2.w + suv.w * s3.w);
    *reinterpret_cast<uint2*>(out + off) = pack4bf(ex, ey, ez, ew);
}

// ============================================================================
//          FAST GEMM KERNELS (all-bf16 operands, async staged, pipelined)
// ============================================================================

__global__ __launch_bounds__(FTHREADS)
void gateup_fast(const u16* __restrict__ xbf,   // [T, H] bf16
                 const u16* __restrict__ geff,  // [I, H] bf16
                 const u16* __restrict__ ueff,  // [I, H] bf16
                 u16* __restrict__ dmat)        // [T, I] bf16
{
    __shared__ __align__(16) u16 sX[2][FBM * SK];
    __shared__ __align__(16) u16 sG[2][BN * SK];
    __shared__ __align__(16) u16 sU[2][BN * SK];

    const int n0 = blockIdx.x * BN;
    const int m0 = blockIdx.y * FBM;

    const int t    = threadIdx.x;
    const int lane = t & 31;
    const int wid  = t >> 5;            // 0..15
    const int wm   = (wid & 3) * 32;    // wave M offset: 0..96
    const int wn   = (wid >> 2) * 32;   // wave N offset: 0..96

    // staging map: 4 threads per row, 16B each; 512 threads cover 128 rows
    const int cx = t & 3;
    const int rx = t >> 2;              // 0..127

    u32 vx = (u32)(((m0 + rx) * H_DIM + cx * 8) * 2);
    u32 vg = (u32)(((n0 + rx) * H_DIM + cx * 8) * 2);
    u32 lX[2], lG[2], lU[2];
#pragma unroll
    for (int b = 0; b < 2; ++b) {
        lX[b] = (u32)(size_t)(&sX[b][rx * SK + cx * 8]);
        lG[b] = (u32)(size_t)(&sG[b][rx * SK + cx * 8]);
        lU[b] = (u32)(size_t)(&sU[b][rx * SK + cx * 8]);
    }

    const v8f vzero = {0.f, 0.f, 0.f, 0.f, 0.f, 0.f, 0.f, 0.f};
    v8f accg[2][2], accu[2][2];
#pragma unroll
    for (int i = 0; i < 2; ++i)
#pragma unroll
        for (int j = 0; j < 2; ++j) { accg[i][j] = vzero; accu[i][j] = vzero; }

    auto issue = [&](int b, u32 kb) {
        async_b128(lX[b], xbf,  vx + kb);
        async_b128(lG[b], geff, vg + kb);
        async_b128(lU[b], ueff, vg + kb);
    };
    auto compute = [&](const u16* sxp, const u16* sgp, const u16* sup) {
        v16bf a0  = load_frag(sxp, wm + 0,  lane);
        v16bf a1  = load_frag(sxp, wm + 16, lane);
        v16bf bg0 = load_frag(sgp, wn + 0,  lane);
        v16bf bg1 = load_frag(sgp, wn + 16, lane);
        v16bf bu0 = load_frag(sup, wn + 0,  lane);
        v16bf bu1 = load_frag(sup, wn + 16, lane);
        accg[0][0] = __builtin_amdgcn_wmma_f32_16x16x32_bf16(false, a0, false, bg0, (short)0, accg[0][0], false, false);
        accg[0][1] = __builtin_amdgcn_wmma_f32_16x16x32_bf16(false, a0, false, bg1, (short)0, accg[0][1], false, false);
        accg[1][0] = __builtin_amdgcn_wmma_f32_16x16x32_bf16(false, a1, false, bg0, (short)0, accg[1][0], false, false);
        accg[1][1] = __builtin_amdgcn_wmma_f32_16x16x32_bf16(false, a1, false, bg1, (short)0, accg[1][1], false, false);
        accu[0][0] = __builtin_amdgcn_wmma_f32_16x16x32_bf16(false, a0, false, bu0, (short)0, accu[0][0], false, false);
        accu[0][1] = __builtin_amdgcn_wmma_f32_16x16x32_bf16(false, a0, false, bu1, (short)0, accu[0][1], false, false);
        accu[1][0] = __builtin_amdgcn_wmma_f32_16x16x32_bf16(false, a1, false, bu0, (short)0, accu[1][0], false, false);
        accu[1][1] = __builtin_amdgcn_wmma_f32_16x16x32_bf16(false, a1, false, bu1, (short)0, accu[1][1], false, false);
    };

    const int NC = H_DIM / BK;   // 128 (even)
    issue(0, 0);
    for (int c = 0; c < NC; c += 2) {
        // ---- chunk c in buffer 0; prefetch c+1 into buffer 1
        issue(1, (u32)(c + 1) << 6);
        wait_async_le<3>();
        __syncthreads();
        compute(sX[0], sG[0], sU[0]);
        __syncthreads();
        // ---- chunk c+1 in buffer 1; prefetch c+2 into buffer 0
        if (c + 2 < NC) { issue(0, (u32)(c + 2) << 6); wait_async_le<3>(); }
        else            { wait_async_le<0>(); }
        __syncthreads();
        compute(sX[1], sG[1], sU[1]);
        __syncthreads();
    }

#pragma unroll
    for (int mi = 0; mi < 2; ++mi)
#pragma unroll
        for (int ni = 0; ni < 2; ++ni)
#pragma unroll
            for (int g = 0; g < 8; ++g) {
                int m = m0 + wm + mi * 16 + g + ((lane >> 4) << 3);
                int n = n0 + wn + ni * 16 + (lane & 15);
                float gv = accg[mi][ni][g];
                float uv = accu[mi][ni][g];
                float dv = (gv / (1.0f + __expf(-gv))) * uv;
                dmat[(size_t)m * I_DIM + n] = f2bf(dv);
            }
}

__global__ __launch_bounds__(FTHREADS)
void down_fast(const u16* __restrict__ dmat,  // [T, I] bf16
               const u16* __restrict__ deff,  // [H, I] bf16
               float* __restrict__ out)       // [T, H]
{
    __shared__ __align__(16) u16 sX[2][FBM * SK];
    __shared__ __align__(16) u16 sW[2][BN * SK];

    const int n0 = blockIdx.x * BN;
    const int m0 = blockIdx.y * FBM;

    const int t    = threadIdx.x;
    const int lane = t & 31;
    const int wid  = t >> 5;
    const int wm   = (wid & 3) * 32;
    const int wn   = (wid >> 2) * 32;

    const int cx = t & 3;
    const int rx = t >> 2;

    u32 vx = (u32)(((m0 + rx) * I_DIM + cx * 8) * 2);
    u32 vw = (u32)(((n0 + rx) * I_DIM + cx * 8) * 2);
    u32 lX[2], lW[2];
#pragma unroll
    for (int b = 0; b < 2; ++b) {
        lX[b] = (u32)(size_t)(&sX[b][rx * SK + cx * 8]);
        lW[b] = (u32)(size_t)(&sW[b][rx * SK + cx * 8]);
    }

    const v8f vzero = {0.f, 0.f, 0.f, 0.f, 0.f, 0.f, 0.f, 0.f};
    v8f acc[2][2];
#pragma unroll
    for (int i = 0; i < 2; ++i)
#pragma unroll
        for (int j = 0; j < 2; ++j) acc[i][j] = vzero;

    auto issue = [&](int b, u32 kb) {
        async_b128(lX[b], dmat, vx + kb);
        async_b128(lW[b], deff, vw + kb);
    };
    auto compute = [&](const u16* sxp, const u16* swp) {
        v16bf a0 = load_frag(sxp, wm + 0,  lane);
        v16bf a1 = load_frag(sxp, wm + 16, lane);
        v16bf b0 = load_frag(swp, wn + 0,  lane);
        v16bf b1 = load_frag(swp, wn + 16, lane);
        acc[0][0] = __builtin_amdgcn_wmma_f32_16x16x32_bf16(false, a0, false, b0, (short)0, acc[0][0], false, false);
        acc[0][1] = __builtin_amdgcn_wmma_f32_16x16x32_bf16(false, a0, false, b1, (short)0, acc[0][1], false, false);
        acc[1][0] = __builtin_amdgcn_wmma_f32_16x16x32_bf16(false, a1, false, b0, (short)0, acc[1][0], false, false);
        acc[1][1] = __builtin_amdgcn_wmma_f32_16x16x32_bf16(false, a1, false, b1, (short)0, acc[1][1], false, false);
    };

    const int NC = I_DIM / BK;   // 448 (even)
    issue(0, 0);
    for (int c = 0; c < NC; c += 2) {
        issue(1, (u32)(c + 1) << 6);
        wait_async_le<2>();
        __syncthreads();
        compute(sX[0], sW[0]);
        __syncthreads();
        if (c + 2 < NC) { issue(0, (u32)(c + 2) << 6); wait_async_le<2>(); }
        else            { wait_async_le<0>(); }
        __syncthreads();
        compute(sX[1], sW[1]);
        __syncthreads();
    }

#pragma unroll
    for (int mi = 0; mi < 2; ++mi)
#pragma unroll
        for (int ni = 0; ni < 2; ++ni)
#pragma unroll
            for (int g = 0; g < 8; ++g) {
                int m = m0 + wm + mi * 16 + g + ((lane >> 4) << 3);
                int n = n0 + wn + ni * 16 + (lane & 15);
                out[(size_t)m * H_DIM + n] = acc[mi][ni][g];
            }
}

// ============================================================================
//        FALLBACK (fused) KERNELS — used if ws is too small
// ============================================================================

__global__ __launch_bounds__(NTHREADS)
void qwen_gateup_swiglu(const float* __restrict__ x,
                        const float* __restrict__ gw,
                        const float* __restrict__ gsu,
                        const float* __restrict__ gsv,
                        const float* __restrict__ uw,
                        const float* __restrict__ usu,
                        const float* __restrict__ usv,
                        u16* __restrict__ dmat)
{
    __shared__ __align__(16) u16 sX[BM * SK];
    __shared__ __align__(16) u16 sG[BN * SK];
    __shared__ __align__(16) u16 sU[BN * SK];

    const int n0 = blockIdx.x * BN;
    const int m0 = blockIdx.y * BM;
    const int t    = threadIdx.x;
    const int lane = t & 31;
    const int wid  = t >> 5;
    const int wm   = (wid & 1) * 32;
    const int wn   = (wid >> 1) * 32;
    const int kq = (t & 7) * 4;
    const int rr = t >> 3;

    float4 gsu4[4], usu4[4];
#pragma unroll
    for (int p = 0; p < 4; ++p) {
        int o = n0 + rr + 32 * p;
        gsu4[p] = *reinterpret_cast<const float4*>(gsu + (size_t)o * RANK);
        usu4[p] = *reinterpret_cast<const float4*>(usu + (size_t)o * RANK);
    }

    const v8f vzero = {0.f, 0.f, 0.f, 0.f, 0.f, 0.f, 0.f, 0.f};
    v8f accg[2][2], accu[2][2];
#pragma unroll
    for (int i = 0; i < 2; ++i)
#pragma unroll
        for (int j = 0; j < 2; ++j) { accg[i][j] = vzero; accu[i][j] = vzero; }

    for (int k0 = 0; k0 < H_DIM; k0 += BK) {
#pragma unroll
        for (int p = 0; p < 2; ++p) {
            int row = rr + 32 * p;
            const float4 xv =
                *reinterpret_cast<const float4*>(x + (size_t)(m0 + row) * H_DIM + k0 + kq);
            *reinterpret_cast<uint2*>(sX + row * SK + kq) = pack4bf(xv.x, xv.y, xv.z, xv.w);
        }
        float4 gsvv[RANK], usvv[RANK];
#pragma unroll
        for (int r = 0; r < RANK; ++r) {
            gsvv[r] = *reinterpret_cast<const float4*>(gsv + (size_t)r * H_DIM + k0 + kq);
            usvv[r] = *reinterpret_cast<const float4*>(usv + (size_t)r * H_DIM + k0 + kq);
        }
#pragma unroll
        for (int p = 0; p < 4; ++p) {
            int row = rr + 32 * p;
            size_t off = (size_t)(n0 + row) * H_DIM + k0 + kq;
            float4 wg = *reinterpret_cast<const float4*>(gw + off);
            float4 sg;
            sg.x = gsu4[p].x * gsvv[0].x + gsu4[p].y * gsvv[1].x + gsu4[p].z * gsvv[2].x + gsu4[p].w * gsvv[3].x;
            sg.y = gsu4[p].x * gsvv[0].y + gsu4[p].y * gsvv[1].y + gsu4[p].z * gsvv[2].y + gsu4[p].w * gsvv[3].y;
            sg.z = gsu4[p].x * gsvv[0].z + gsu4[p].y * gsvv[1].z + gsu4[p].z * gsvv[2].z + gsu4[p].w * gsvv[3].z;
            sg.w = gsu4[p].x * gsvv[0].w + gsu4[p].y * gsvv[1].w + gsu4[p].z * gsvv[2].w + gsu4[p].w * gsvv[3].w;
            *reinterpret_cast<uint2*>(sG + row * SK + kq) =
                pack4bf(wg.x * sg.x, wg.y * sg.y, wg.z * sg.z, wg.w * sg.w);

            float4 wu = *reinterpret_cast<const float4*>(uw + off);
            float4 su_;
            su_.x = usu4[p].x * usvv[0].x + usu4[p].y * usvv[1].x + usu4[p].z * usvv[2].x + usu4[p].w * usvv[3].x;
            su_.y = usu4[p].x * usvv[0].y + usu4[p].y * usvv[1].y + usu4[p].z * usvv[2].y + usu4[p].w * usvv[3].y;
            su_.z = usu4[p].x * usvv[0].z + usu4[p].y * usvv[1].z + usu4[p].z * usvv[2].z + usu4[p].w * usvv[3].z;
            su_.w = usu4[p].x * usvv[0].w + usu4[p].y * usvv[1].w + usu4[p].z * usvv[2].w + usu4[p].w * usvv[3].w;
            *reinterpret_cast<uint2*>(sU + row * SK + kq) =
                pack4bf(wu.x * su_.x, wu.y * su_.y, wu.z * su_.z, wu.w * su_.w);
        }
        __syncthreads();

        v16bf a0  = load_frag(sX, wm + 0,  lane);
        v16bf a1  = load_frag(sX, wm + 16, lane);
        v16bf bg0 = load_frag(sG, wn + 0,  lane);
        v16bf bg1 = load_frag(sG, wn + 16, lane);
        v16bf bu0 = load_frag(sU, wn + 0,  lane);
        v16bf bu1 = load_frag(sU, wn + 16, lane);

        accg[0][0] = __builtin_amdgcn_wmma_f32_16x16x32_bf16(false, a0, false, bg0, (short)0, accg[0][0], false, false);
        accg[0][1] = __builtin_amdgcn_wmma_f32_16x16x32_bf16(false, a0, false, bg1, (short)0, accg[0][1], false, false);
        accg[1][0] = __builtin_amdgcn_wmma_f32_16x16x32_bf16(false, a1, false, bg0, (short)0, accg[1][0], false, false);
        accg[1][1] = __builtin_amdgcn_wmma_f32_16x16x32_bf16(false, a1, false, bg1, (short)0, accg[1][1], false, false);
        accu[0][0] = __builtin_amdgcn_wmma_f32_16x16x32_bf16(false, a0, false, bu0, (short)0, accu[0][0], false, false);
        accu[0][1] = __builtin_amdgcn_wmma_f32_16x16x32_bf16(false, a0, false, bu1, (short)0, accu[0][1], false, false);
        accu[1][0] = __builtin_amdgcn_wmma_f32_16x16x32_bf16(false, a1, false, bu0, (short)0, accu[1][0], false, false);
        accu[1][1] = __builtin_amdgcn_wmma_f32_16x16x32_bf16(false, a1, false, bu1, (short)0, accu[1][1], false, false);

        __syncthreads();
    }

#pragma unroll
    for (int mi = 0; mi < 2; ++mi)
#pragma unroll
        for (int ni = 0; ni < 2; ++ni)
#pragma unroll
            for (int g = 0; g < 8; ++g) {
                int m = m0 + wm + mi * 16 + g + ((lane >> 4) << 3);
                int n = n0 + wn + ni * 16 + (lane & 15);
                float gv = accg[mi][ni][g];
                float uv = accu[mi][ni][g];
                float dv = (gv / (1.0f + __expf(-gv))) * uv;
                dmat[(size_t)m * I_DIM + n] = f2bf(dv);
            }
}

__global__ __launch_bounds__(NTHREADS)
void qwen_down(const u16*  __restrict__ dmat,
               const float* __restrict__ dw,
               const float* __restrict__ dsu,
               const float* __restrict__ dsv,
               float* __restrict__ out)
{
    __shared__ __align__(16) u16 sX[BM * SK];
    __shared__ __align__(16) u16 sW[BN * SK];

    const int n0 = blockIdx.x * BN;
    const int m0 = blockIdx.y * BM;
    const int t    = threadIdx.x;
    const int lane = t & 31;
    const int wid  = t >> 5;
    const int wm   = (wid & 1) * 32;
    const int wn   = (wid >> 1) * 32;
    const int kq = (t & 7) * 4;
    const int rr = t >> 3;

    float4 dsu4[4];
#pragma unroll
    for (int p = 0; p < 4; ++p) {
        int o = n0 + rr + 32 * p;
        dsu4[p] = *reinterpret_cast<const float4*>(dsu + (size_t)o * RANK);
    }

    const v8f vzero = {0.f, 0.f, 0.f, 0.f, 0.f, 0.f, 0.f, 0.f};
    v8f acc[2][2];
#pragma unroll
    for (int i = 0; i < 2; ++i)
#pragma unroll
        for (int j = 0; j < 2; ++j) acc[i][j] = vzero;

    for (int k0 = 0; k0 < I_DIM; k0 += BK) {
#pragma unroll
        for (int p = 0; p < 2; ++p) {
            int row = rr + 32 * p;
            uint2 v = *reinterpret_cast<const uint2*>(dmat + (size_t)(m0 + row) * I_DIM + k0 + kq);
            *reinterpret_cast<uint2*>(sX + row * SK + kq) = v;
        }
        float4 svv[RANK];
#pragma unroll
        for (int r = 0; r < RANK; ++r)
            svv[r] = *reinterpret_cast<const float4*>(dsv + (size_t)r * I_DIM + k0 + kq);
#pragma unroll
        for (int p = 0; p < 4; ++p) {
            int row = rr + 32 * p;
            size_t off = (size_t)(n0 + row) * I_DIM + k0 + kq;
            float4 wv = *reinterpret_cast<const float4*>(dw + off);
            float4 s;
            s.x = dsu4[p].x * svv[0].x + dsu4[p].y * svv[1].x + dsu4[p].z * svv[2].x + dsu4[p].w * svv[3].x;
            s.y = dsu4[p].x * svv[0].y + dsu4[p].y * svv[1].y + dsu4[p].z * svv[2].y + dsu4[p].w * svv[3].y;
            s.z = dsu4[p].x * svv[0].z + dsu4[p].y * svv[1].z + dsu4[p].z * svv[2].z + dsu4[p].w * svv[3].z;
            s.w = dsu4[p].x * svv[0].w + dsu4[p].y * svv[1].w + dsu4[p].z * svv[2].w + dsu4[p].w * svv[3].w;
            *reinterpret_cast<uint2*>(sW + row * SK + kq) =
                pack4bf(wv.x * s.x, wv.y * s.y, wv.z * s.z, wv.w * s.w);
        }
        __syncthreads();

        v16bf a0 = load_frag(sX, wm + 0,  lane);
        v16bf a1 = load_frag(sX, wm + 16, lane);
        v16bf b0 = load_frag(sW, wn + 0,  lane);
        v16bf b1 = load_frag(sW, wn + 16, lane);

        acc[0][0] = __builtin_amdgcn_wmma_f32_16x16x32_bf16(false, a0, false, b0, (short)0, acc[0][0], false, false);
        acc[0][1] = __builtin_amdgcn_wmma_f32_16x16x32_bf16(false, a0, false, b1, (short)0, acc[0][1], false, false);
        acc[1][0] = __builtin_amdgcn_wmma_f32_16x16x32_bf16(false, a1, false, b0, (short)0, acc[1][0], false, false);
        acc[1][1] = __builtin_amdgcn_wmma_f32_16x16x32_bf16(false, a1, false, b1, (short)0, acc[1][1], false, false);

        __syncthreads();
    }

#pragma unroll
    for (int mi = 0; mi < 2; ++mi)
#pragma unroll
        for (int ni = 0; ni < 2; ++ni)
#pragma unroll
            for (int g = 0; g < 8; ++g) {
                int m = m0 + wm + mi * 16 + g + ((lane >> 4) << 3);
                int n = n0 + wn + ni * 16 + (lane & 15);
                out[(size_t)m * H_DIM + n] = acc[mi][ni][g];
            }
}

// ============================================================================

extern "C" void kernel_launch(void* const* d_in, const int* in_sizes, int n_in,
                              void* d_out, int out_size, void* d_ws, size_t ws_size,
                              hipStream_t stream) {
    (void)in_sizes; (void)n_in; (void)out_size;
    const float* x   = (const float*)d_in[0];
    const float* gw  = (const float*)d_in[1];
    const float* gsu = (const float*)d_in[2];
    const float* gsv = (const float*)d_in[3];
    const float* uw  = (const float*)d_in[4];
    const float* usu = (const float*)d_in[5];
    const float* usv = (const float*)d_in[6];
    const float* dw  = (const float*)d_in[7];
    const float* dsu = (const float*)d_in[8];
    const float* dsv = (const float*)d_in[9];
    float* out = (float*)d_out;

    const size_t XBF  = (size_t)T_DIM * H_DIM * 2;   // 8.4 MB
    const size_t WEFF = (size_t)I_DIM * H_DIM * 2;   // 117.4 MB each
    const size_t DMAT = (size_t)T_DIM * I_DIM * 2;   // 29.4 MB
    const size_t NEED = XBF + 3 * WEFF + DMAT;       // ~372 MB

    char* ws = (char*)d_ws;
    if (ws_size >= NEED) {
        u16* xbf  = (u16*)ws;
        u16* geff = (u16*)(ws + XBF);
        u16* ueff = (u16*)(ws + XBF + WEFF);
        u16* deff = (u16*)(ws + XBF + 2 * WEFF);
        u16* dmat = (u16*)(ws + XBF + 3 * WEFF);

        cvt_x_kernel<<<(T_DIM * H_DIM) / 1024, 256, 0, stream>>>(x, xbf);
        scale_cvt_kernel<<<dim3(H_DIM / 1024, I_DIM), 256, 0, stream>>>(gw, gsu, gsv, geff, H_DIM);
        scale_cvt_kernel<<<dim3(H_DIM / 1024, I_DIM), 256, 0, stream>>>(uw, usu, usv, ueff, H_DIM);
        scale_cvt_kernel<<<dim3(I_DIM / 1024, H_DIM), 256, 0, stream>>>(dw, dsu, dsv, deff, I_DIM);

        dim3 g1(I_DIM / BN, T_DIM / FBM);            // (112, 8)
        gateup_fast<<<g1, FTHREADS, 0, stream>>>(xbf, geff, ueff, dmat);
        dim3 g2(H_DIM / BN, T_DIM / FBM);            // (32, 8)
        down_fast<<<g2, FTHREADS, 0, stream>>>(dmat, deff, out);
    } else {
        u16* dmat = (u16*)ws;                        // needs only 29.4 MB
        dim3 g1(I_DIM / BN, T_DIM / BM);
        qwen_gateup_swiglu<<<g1, NTHREADS, 0, stream>>>(x, gw, gsu, gsv, uw, usu, usv, dmat);
        dim3 g2(H_DIM / BN, T_DIM / BM);
        qwen_down<<<g2, NTHREADS, 0, stream>>>(dmat, dw, dsu, dsv, out);
    }
}
